// ImprovedEEGModel_82712480187064
// MI455X (gfx1250) — compile-verified, matrix operand order checked by reading, not run on hardware
//
#include <hip/hip_runtime.h>
#include <hip/hip_bf16.h>
#include <math.h>

// ---------------------------------------------------------------------------
// ImprovedEEGModel fused pipeline for gfx1250 (MI455X, wave32, WMMA).
//
// Kernel 1 (init):  build 128x64 f16 DFT matrix (Hann window + detrend
//                   correction folded in) directly in WMMA B-operand lane
//                   layout: coef[nt][ks][lane][j], j -> K = 8*(lane>>4) +
//                   (j&7) + ((j>>3)<<4) + 32*ks, N = nt*16 + (lane&15).
//                   N 0..29 = cos bins 1..30 (+0.25 detrend corr at bin 1),
//                   N 32..61 = -sin bins 1..30, else 0.  K >= 100 zero-padded.
// Kernel 2 (welch): one pass over x. 16 rows/block (128 thr = 4 waves).
//                   Row tile -> LDS (stride 204 floats, conflict-free),
//                   mean/max reduction, then 3 segments x 4 ksteps of
//                   v_wmma_f32_16x16x32_f16 per wave (wave = N-tile).
//                   |X|^2 -> one-sided scaled PSD -> 4 band means.
// Kernel 3 (head):  attention MLP + backbone + freq encoder + head, fp32.
// ---------------------------------------------------------------------------

typedef __attribute__((ext_vector_type(16))) _Float16 v16h;
typedef __attribute__((ext_vector_type(8)))  float    v8f;

#define PI_F 3.14159265358979323846f

constexpr int B_   = 2048;
constexpr int C_   = 129;
constexpr int T_   = 200;
constexpr int R_   = B_ * C_;        // 264192 rows
constexpr int TILE = 16;             // rows per workgroup (WMMA M)
constexpr int NTIL = R_ / TILE;      // 16512 workgroups
constexpr int XSTR = 204;            // LDS row stride in floats (16B aligned, conflict-free)
constexpr int RSTR = 68;             // result row stride in floats (conflict-free)
constexpr int NCOEF = 4 * 4 * 32 * 16;  // ntile * kstep * lane * elem = 8192 f16

// ---------------------------------------------------------------------------
// DFT coefficient table in native WMMA B-operand layout.
__global__ void eeg_init_coef(_Float16* __restrict__ coef) {
    int t = blockIdx.x * blockDim.x + threadIdx.x;
    if (t >= NCOEF) return;
    int j    = t & 15;
    int lane = (t >> 4) & 31;
    int ks   = (t >> 9) & 3;
    int nt   = t >> 11;
    int q = ks * 32 + ((lane >> 4) << 3) + (j & 7) + ((j >> 3) << 4);  // K index
    int n = nt * 16 + (lane & 15);                                     // N index
    float v = 0.0f;
    if (q < 100) {
        float w = 0.5f - 0.5f * cosf(2.0f * PI_F * (float)q * 0.01f);  // periodic hann
        if (n < 30) {
            int bin = n + 1;
            int ph = (bin * q) % 100;                                  // exact phase reduction
            v = w * cosf(2.0f * PI_F * (float)ph * 0.01f);
            if (bin == 1) v += 0.25f;   // detrend: -W_1/N = -(-25)/100
        } else if (n >= 32 && n < 62) {
            int bin = n - 31;
            int ph = (bin * q) % 100;
            v = -w * sinf(2.0f * PI_F * (float)ph * 0.01f);
        }
    }
    coef[t] = (_Float16)v;
}

// ---------------------------------------------------------------------------
// Fused pass over x: mean, max, Welch band powers (WMMA DFT).
__global__ __launch_bounds__(128)
void eeg_welch_stats(const float* __restrict__ x,
                     const _Float16* __restrict__ coef,
                     float* __restrict__ avg_out,
                     float* __restrict__ mx_out,
                     float* __restrict__ freq_out) {
    __shared__ float sm_x[TILE * XSTR];        // 13056 B  (16 rows x 200 padded)
    __shared__ float sm_res[3 * TILE * RSTR];  // 13056 B  (3 segs x 16 x 64 padded)
    __shared__ float sm_psd[TILE * 32];        // 2048 B
    __shared__ float sm_sum[TILE * 8];
    __shared__ float sm_max[TILE * 8];

    const int tid  = threadIdx.x;
    const int wave = tid >> 5;     // 0..3 -> N-tile
    const int lane = tid & 31;
    const size_t tilebase = (size_t)blockIdx.x * (TILE * T_);

    // ---- stage x tile into LDS (contiguous 12.8 KB chunk) -----------------
    const float4* gx = (const float4*)(x + tilebase);
    for (int g = tid; g < TILE * T_ / 4; g += 128) {
        float4 v = gx[g];
        int row = g / 50, q4 = g % 50;                 // 50 float4 per row
        float4* dst = (float4*)(sm_x + row * XSTR);
        dst[q4] = v;
    }
    __syncthreads();

    // ---- mean / max partial reduction (8 partials per row) ----------------
    {
        int row = tid >> 3, part = tid & 7;
        const float* rp = sm_x + row * XSTR + part * 25;
        float s = 0.0f, m = -3.4e38f;
        #pragma unroll
        for (int i = 0; i < 25; ++i) { float v = rp[i]; s += v; m = fmaxf(m, v); }
        sm_sum[row * 8 + part] = s;
        sm_max[row * 8 + part] = m;
    }

    // ---- load this wave's DFT B-operands (L2-resident table) --------------
    v16h bq[4];
    {
        const v16h* cp = (const v16h*)coef;
        #pragma unroll
        for (int ks = 0; ks < 4; ++ks)
            bq[ks] = cp[(wave * 4 + ks) * 32 + lane];
    }
    __syncthreads();

    if (tid < TILE) {
        float s = 0.0f, m = -3.4e38f;
        #pragma unroll
        for (int p = 0; p < 8; ++p) { s += sm_sum[tid * 8 + p]; m = fmaxf(m, sm_max[tid * 8 + p]); }
        size_t r = (size_t)blockIdx.x * TILE + tid;
        avg_out[r] = s * (1.0f / (float)T_);
        mx_out[r]  = m;
    }

    // ---- WMMA DFT: 3 segments x K=128 (padded), this wave's N-tile --------
    const int m    = lane & 15;
    const int base = (lane >> 4) << 3;
    const float* rowp = sm_x + m * XSTR;
    for (int s = 0; s < 3; ++s) {
        v8f acc = {};
        #pragma unroll
        for (int ks = 0; ks < 4; ++ks) {
            v16h a;
            int q0 = ks * 32 + base;
            #pragma unroll
            for (int j = 0; j < 16; ++j) {
                int q = q0 + (j & 7) + ((j >> 3) << 4);        // segment-local sample
                a[j] = (_Float16)((q < 100) ? rowp[s * 50 + q] : 0.0f);
            }
            acc = __builtin_amdgcn_wmma_f32_16x16x32_f16(
                      /*neg_a=*/false, a, /*neg_b=*/false, bq[ks],
                      /*c_mod=*/(short)0, acc, /*reuse_a=*/false, /*reuse_b=*/false);
        }
        #pragma unroll
        for (int v = 0; v < 8; ++v) {
            int mm = v + ((lane >> 4) << 3);                   // C/D layout: M = v + 8*(lane/16)
            sm_res[s * (TILE * RSTR) + mm * RSTR + wave * 16 + (lane & 15)] = acc[v];
        }
    }
    __syncthreads();

    // ---- per-bin one-sided PSD averaged over 3 segments -------------------
    // scale = 2 / (3 * SFREQ * sum(win^2)) = 2 / (3 * 100 * 37.5)
    for (int job = tid; job < TILE * 30; job += 128) {
        int row = job / 30, k = job % 30 + 1;
        float p = 0.0f;
        #pragma unroll
        for (int s = 0; s < 3; ++s) {
            float cr = sm_res[s * (TILE * RSTR) + row * RSTR + (k - 1)];
            float ci = sm_res[s * (TILE * RSTR) + row * RSTR + (k + 31)];
            p += cr * cr + ci * ci;
        }
        sm_psd[row * 32 + (k - 1)] = p * (2.0f / 11250.0f);
    }
    __syncthreads();

    // ---- band means: k in [1,4],[4,8],[8,13],[13,30] (inclusive) ----------
    if (tid < TILE * 4) {
        const int lo[4]  = {0, 3, 7, 12};      // (k-1) start
        const int cnt[4] = {4, 5, 6, 18};
        int row = tid >> 2, band = tid & 3;
        float s = 0.0f;
        for (int i = 0; i < cnt[band]; ++i) s += sm_psd[row * 32 + lo[band] + i];
        size_t r = (size_t)blockIdx.x * TILE + row;
        freq_out[r * 4 + band] = s / (float)cnt[band];
    }
}

// ---------------------------------------------------------------------------
// Attention + backbone + freq encoder + head (tiny; fp32 VALU, LDS staged).
__global__ __launch_bounds__(256)
void eeg_head(const float* __restrict__ avg_g, const float* __restrict__ mx_g,
              const float* __restrict__ freq_g,
              const float* __restrict__ w_att1, const float* __restrict__ w_att2,
              const float* __restrict__ w_bb,   const float* __restrict__ b_bb,
              const float* __restrict__ w_fe1,  const float* __restrict__ b_fe1,
              const float* __restrict__ w_fe2,  const float* __restrict__ b_fe2,
              const float* __restrict__ w_h1,   const float* __restrict__ b_h1,
              const float* __restrict__ w_h2,   const float* __restrict__ b_h2,
              const float* __restrict__ w_h3,   const float* __restrict__ b_h3,
              float* __restrict__ out) {
    __shared__ float s_avg[16 * 132];   // avg rows
    __shared__ float s_mx [16 * 132];   // mx rows, later reused for pooled
    __shared__ float s_u  [16 * 16];    // relu(avg@W1)+relu(mx@W1)
    __shared__ float s_h  [16 * 128];   // concat [time_feat | f2]
    __shared__ float s_t  [16 * 128];   // temp: f1, later h1
    __shared__ float s_h2 [16 * 64];

    const int tid = threadIdx.x;
    const int bg0 = blockIdx.x * 16;    // first batch row of this block

    // stage avg / mx rows
    for (int job = tid; job < 16 * C_; job += 256) {
        int m = job / C_, c = job % C_;
        s_avg[m * 132 + c] = avg_g[(size_t)(bg0 + m) * C_ + c];
        s_mx [m * 132 + c] = mx_g [(size_t)(bg0 + m) * C_ + c];
    }
    __syncthreads();

    // u = relu(avg @ w_att1) + relu(mx @ w_att1)      (16 x 16)
    for (int job = tid; job < 16 * 16; job += 256) {
        int m = job >> 4, r = job & 15;
        float da = 0.0f, dm = 0.0f;
        for (int c = 0; c < C_; ++c) {
            float w = w_att1[c * 16 + r];
            da += s_avg[m * 132 + c] * w;
            dm += s_mx [m * 132 + c] * w;
        }
        s_u[m * 16 + r] = fmaxf(da, 0.0f) + fmaxf(dm, 0.0f);
    }
    __syncthreads();

    // att = sigmoid(u @ w_att2); pooled = avg * att   (16 x 129, into s_mx)
    for (int job = tid; job < 16 * C_; job += 256) {
        int m = job / C_, c = job % C_;
        float z = 0.0f;
        #pragma unroll
        for (int r = 0; r < 16; ++r) z += s_u[m * 16 + r] * w_att2[r * C_ + c];
        float att = 1.0f / (1.0f + expf(-z));
        s_mx[m * 132 + c] = s_avg[m * 132 + c] * att;
    }
    __syncthreads();

    // time_feat = relu(pooled @ w_bb + b_bb) -> s_h[:, 0:64]
    for (int job = tid; job < 16 * 64; job += 256) {
        int m = job >> 6, n = job & 63;
        float d = b_bb[n];
        for (int c = 0; c < C_; ++c) d += s_mx[m * 132 + c] * w_bb[c * 64 + n];
        s_h[m * 128 + n] = fmaxf(d, 0.0f);
    }
    // f1 = relu(freq @ w_fe1 + b_fe1) -> s_t          (independent of above)
    for (int job = tid; job < 16 * 128; job += 256) {
        int m = job >> 7, n = job & 127;
        float d = b_fe1[n];
        const float* fr = freq_g + (size_t)(bg0 + m) * (C_ * 4);
        for (int k = 0; k < C_ * 4; ++k) d += fr[k] * w_fe1[k * 128 + n];
        s_t[m * 128 + n] = fmaxf(d, 0.0f);
    }
    __syncthreads();

    // f2 = relu(f1 @ w_fe2 + b_fe2) -> s_h[:, 64:128]
    for (int job = tid; job < 16 * 64; job += 256) {
        int m = job >> 6, n = job & 63;
        float d = b_fe2[n];
        #pragma unroll 8
        for (int k = 0; k < 128; ++k) d += s_t[m * 128 + k] * w_fe2[k * 64 + n];
        s_h[m * 128 + 64 + n] = fmaxf(d, 0.0f);
    }
    __syncthreads();

    // h1 = relu(h @ w_h1 + b_h1) -> s_t
    for (int job = tid; job < 16 * 128; job += 256) {
        int m = job >> 7, n = job & 127;
        float d = b_h1[n];
        #pragma unroll 8
        for (int k = 0; k < 128; ++k) d += s_h[m * 128 + k] * w_h1[k * 128 + n];
        s_t[m * 128 + n] = fmaxf(d, 0.0f);
    }
    __syncthreads();

    // h2 = relu(h1 @ w_h2 + b_h2) -> s_h2
    for (int job = tid; job < 16 * 64; job += 256) {
        int m = job >> 6, n = job & 63;
        float d = b_h2[n];
        #pragma unroll 8
        for (int k = 0; k < 128; ++k) d += s_t[m * 128 + k] * w_h2[k * 64 + n];
        s_h2[m * 64 + n] = fmaxf(d, 0.0f);
    }
    __syncthreads();

    // out = h2 @ w_h3 + b_h3
    if (tid < 16) {
        float d = b_h3[0];
        #pragma unroll 8
        for (int k = 0; k < 64; ++k) d += s_h2[tid * 64 + k] * w_h3[k];
        out[bg0 + tid] = d;
    }
}

// ---------------------------------------------------------------------------
extern "C" void kernel_launch(void* const* d_in, const int* in_sizes, int n_in,
                              void* d_out, int out_size, void* d_ws, size_t ws_size,
                              hipStream_t stream) {
    (void)in_sizes; (void)n_in; (void)out_size; (void)ws_size;
    const float* x      = (const float*)d_in[0];
    const float* w_att1 = (const float*)d_in[1];
    const float* w_att2 = (const float*)d_in[2];
    const float* w_bb   = (const float*)d_in[3];
    const float* b_bb   = (const float*)d_in[4];
    const float* w_fe1  = (const float*)d_in[5];
    const float* b_fe1  = (const float*)d_in[6];
    const float* w_fe2  = (const float*)d_in[7];
    const float* b_fe2  = (const float*)d_in[8];
    const float* w_h1   = (const float*)d_in[9];
    const float* b_h1   = (const float*)d_in[10];
    const float* w_h2   = (const float*)d_in[11];
    const float* b_h2   = (const float*)d_in[12];
    const float* w_h3   = (const float*)d_in[13];
    const float* b_h3   = (const float*)d_in[14];

    // workspace: avg[R] | mx[R] | freq[4R] | coef[8192 f16]
    float* ws   = (float*)d_ws;
    float* avg  = ws;
    float* mx   = ws + (size_t)R_;
    float* freq = ws + 2 * (size_t)R_;
    _Float16* coef = (_Float16*)(ws + 6 * (size_t)R_);

    eeg_init_coef<<<NCOEF / 256, 256, 0, stream>>>(coef);
    eeg_welch_stats<<<NTIL, 128, 0, stream>>>(x, coef, avg, mx, freq);
    eeg_head<<<B_ / 16, 256, 0, stream>>>(avg, mx, freq,
                                          w_att1, w_att2, w_bb, b_bb,
                                          w_fe1, b_fe1, w_fe2, b_fe2,
                                          w_h1, b_h1, w_h2, b_h2, w_h3, b_h3,
                                          (float*)d_out);
}
